// Downsample_Vit_16784732193427
// MI455X (gfx1250) — compile-verified, matrix operand order checked by reading, not run on hardware
//
#include <hip/hip_runtime.h>
#include <hip/hip_bf16.h>

typedef __attribute__((ext_vector_type(16))) _Float16 v16h;
typedef __attribute__((ext_vector_type(8)))  _Float16 v8h;
typedef __attribute__((ext_vector_type(4)))  _Float16 v4h;
typedef __attribute__((ext_vector_type(8)))  float    v8f;
typedef __attribute__((ext_vector_type(4)))  unsigned int u32x4;
typedef __attribute__((ext_vector_type(8)))  int      i32x8;
typedef __attribute__((ext_vector_type(4)))  int      i32x4;
typedef __attribute__((ext_vector_type(2)))  int      v2i;

typedef __attribute__((address_space(1))) void gas_t;
typedef __attribute__((address_space(3))) void las_t;
typedef __attribute__((address_space(1))) v2i  g_v2i;
typedef __attribute__((address_space(3))) v2i  l_v2i;

#ifndef __has_builtin
#define __has_builtin(x) 0
#endif

#if __has_builtin(__builtin_amdgcn_global_load_async_to_lds_b64) && \
    __has_builtin(__builtin_amdgcn_s_wait_asynccnt)
#define USE_ASYNC_LDS 1
#else
#define USE_ASYNC_LDS 0
#endif

#if __has_builtin(__builtin_amdgcn_tensor_load_to_lds) && \
    __has_builtin(__builtin_amdgcn_s_wait_tensorcnt)
#define USE_TDM 1
#ifdef __has_include
#if __has_include(<hip/amd_detail/amd_gfx1250_TDM.h>)
#define TDM_6ARG 1
#else
#define TDM_6ARG 0
#endif
#else
#define TDM_6ARG 0
#endif
#else
#define USE_TDM 0
#endif

#define BB      8
#define C64     64
#define HW      12544      // 112*112
#define CHW     802816     // 64*HW
#define PA      196        // patch attn length
#define MROWS   4096       // C * SHN
#define TGW     56
#define NG      3136       // 56*56 global attn length
#define PS_LD   224        // padded patch prob stride (7*32)
#define PS_ROWS 208        // padded rows (13*16)

#define CAT16(a, b) \
  __builtin_shufflevector(a, b, 0, 1, 2, 3, 4, 5, 6, 7, 8, 9, 10, 11, 12, 13, 14, 15)

// K index of element e (0..15) of a v16h A/B fragment (ISA 16-bit layout)
__device__ __forceinline__ int kof(int e, int l16) {
  int v = e >> 1;
  return ((v < 4) ? 0 : 16) + (l16 ? 8 : 0) + ((v & 3) << 1) + (e & 1);
}

// inverse: given k (0..31) and column n (0..15) -> swizzled (lane, e)
__device__ __forceinline__ int swz_idx(int k, int n) {
  int l16 = (k >> 3) & 1;
  int v = ((k >> 4) & 1) * 4 + ((k >> 1) & 3);
  int e = v * 2 + (k & 1);
  return (l16 * 16 + n) * 16 + e;
}

__device__ __forceinline__ v8f wmma16(v16h a, v16h b, v8f c) {
  return __builtin_amdgcn_wmma_f32_16x16x32_f16(false, a, false, b, (short)0, c,
                                                false, false);
}

__device__ __forceinline__ int iclamp(int v, int lo, int hi) {
  return v < lo ? lo : (v > hi ? hi : v);
}

// ---------------------------------------------------------------- GroupNorm stats
__global__ void k_gnstats(const float* __restrict__ x, float* __restrict__ stats) {
  int b = blockIdx.x;
  const float* xb = x + (size_t)b * CHW;
  float s = 0.f, ss = 0.f;
  for (int i = threadIdx.x; i < CHW; i += blockDim.x) {
    float v = xb[i]; s += v; ss += v * v;
  }
  __shared__ float shs[256], shq[256];
  int t = threadIdx.x;
  shs[t] = s; shq[t] = ss;
  __syncthreads();
  for (int o = 128; o > 0; o >>= 1) {
    if (t < o) { shs[t] += shs[t + o]; shq[t] += shq[t + o]; }
    __syncthreads();
  }
  if (t == 0) {
    float mu  = shs[0] * (1.f / CHW);
    float var = shq[0] * (1.f / CHW) - mu * mu;
    stats[2 * b]     = mu;
    stats[2 * b + 1] = rsqrtf(var + 1e-5f);
  }
}

// ------------------------------------------------- fused GN + QKV (1x1 conv GEMMs)
// grid (196, B), block 128; block tile: M=64 (all), N=64, K=64; wave w -> N subtile
__global__ void k_qkv(const float* __restrict__ x, const float* __restrict__ stats,
                      const float* __restrict__ gnw, const float* __restrict__ gnb,
                      const float* __restrict__ wq, const float* __restrict__ bq,
                      const float* __restrict__ wk, const float* __restrict__ bk,
                      const float* __restrict__ wv, const float* __restrict__ bv,
                      _Float16* __restrict__ qh, _Float16* __restrict__ kh,
                      _Float16* __restrict__ vh) {
  int b = blockIdx.y;
  int n0 = blockIdx.x * 64;
  __shared__ __align__(16) _Float16 lw[3][C64 * C64];
  __shared__ __align__(16) float lxr[C64 * 64];          // raw x tile [c][n]
  __shared__ __align__(32) _Float16 lxs[4][2][32 * 16];  // swizzled B fragments

  for (int i = threadIdx.x; i < C64 * C64; i += blockDim.x) {
    lw[0][i] = (_Float16)wq[i];
    lw[1][i] = (_Float16)wk[i];
    lw[2][i] = (_Float16)wv[i];
  }
  const float* xb = x + (size_t)b * CHW;

#if USE_TDM
  if ((threadIdx.x >> 5) == 0) {
    // Tensor DMA: 2D tile [64 rows][64 cols] of f32 from xb+n0 into lxr
    unsigned long long ga = (unsigned long long)(size_t)(const void*)(xb + n0);
    unsigned int lo = (unsigned int)(size_t)(las_t*)&lxr[0];
    u32x4 g0;
    g0.x = 1u;                                              // count=1
    g0.y = lo;                                              // lds_addr
    g0.z = (unsigned int)ga;                                // global_addr[31:0]
    g0.w = (unsigned int)((ga >> 32) & 0x01FFFFFFu) | 0x80000000u;  // [56:32]|type=2
    i32x8 g1;
    g1[0] = (int)(2u << 16);                                // data_size = 4B
    g1[1] = (int)(((unsigned)HW & 0xFFFFu) << 16);          // tensor_dim0[15:0]
    g1[2] = (int)((((unsigned)HW >> 16) & 0xFFFFu) | (64u << 16));  // dim0hi|dim1lo
    g1[3] = (int)(64u << 16);                               // dim1hi=0 | tile_dim0=64
    g1[4] = (int)64u;                                       // tile_dim1=64, tile_dim2=0
    g1[5] = (int)HW;                                        // dim0_stride lo
    g1[6] = 0;
    g1[7] = 0;
    i32x4 gz = {0, 0, 0, 0};
#if TDM_6ARG
    i32x8 gz8 = {0, 0, 0, 0, 0, 0, 0, 0};
    __builtin_amdgcn_tensor_load_to_lds(g0, g1, gz, gz, gz8, 0);
#else
    __builtin_amdgcn_tensor_load_to_lds(g0, g1, gz, gz, 0);
#endif
    __builtin_amdgcn_s_wait_tensorcnt(0);
  }
  __syncthreads();
#else
  for (int i = threadIdx.x; i < C64 * 64; i += blockDim.x)
    lxr[i] = xb[(size_t)(i >> 6) * HW + n0 + (i & 63)];
  __syncthreads();
#endif

  // normalize + swizzle into fragment order
  float mu = stats[2 * b], rs = stats[2 * b + 1];
  for (int i = threadIdx.x; i < C64 * 64; i += blockDim.x) {
    int c = i >> 6, n = i & 63;
    float v = (lxr[i] - mu) * rs * gnw[c] + gnb[c];
    lxs[n >> 4][c >> 5][swz_idx(c & 31, n & 15)] = (_Float16)v;
  }
  __syncthreads();

  int wave = threadIdx.x >> 5;
  int lane = threadIdx.x & 31, lm = lane & 15, l16 = lane >> 4;

  v16h Bf[2];
  Bf[0] = *(const v16h*)&lxs[wave][0][lane * 16];
  Bf[1] = *(const v16h*)&lxs[wave][1][lane * 16];

  _Float16* outs[3]    = {qh, kh, vh};
  const float* bias[3] = {bq, bk, bv};
#pragma unroll
  for (int mat = 0; mat < 3; mat++) {
#pragma unroll
    for (int mt = 0; mt < 4; mt++) {
      v8f acc = {};
#pragma unroll
      for (int ks = 0; ks < 2; ks++) {
        const v8h* ap =
            (const v8h*)&lw[mat][(mt * 16 + lm) * C64 + ks * 32 + l16 * 8];
        v16h Af = CAT16(ap[0], ap[2]);
        acc = wmma16(Af, Bf[ks], acc);
      }
#pragma unroll
      for (int r = 0; r < 8; r++) {
        int o = mt * 16 + r + (l16 ? 8 : 0);
        float val = acc[r] + bias[mat][o];
        outs[mat][(size_t)b * CHW + (size_t)o * HW + n0 + wave * 16 + lm] =
            (_Float16)val;
      }
    }
  }
}

// ---------------------------------------------------------- 2x2 mean pool 112->56
__global__ void k_pool(const _Float16* __restrict__ q, const _Float16* __restrict__ k,
                       const _Float16* __restrict__ v, _Float16* __restrict__ qg,
                       _Float16* __restrict__ kg, _Float16* __restrict__ vg) {
  int idx = blockIdx.x * blockDim.x + threadIdx.x;
  if (idx >= BB * C64 * NG) return;
  int i = idx % NG;
  int c = (idx / NG) % C64;
  int b = idx / (NG * C64);
  int y = i / TGW, xx = i % TGW;
  size_t p = (size_t)b * CHW + (size_t)c * HW + (size_t)(2 * y) * 112 + 2 * xx;
  float a0 = (float)q[p] + (float)q[p + 1] + (float)q[p + 112] + (float)q[p + 113];
  float a1 = (float)k[p] + (float)k[p + 1] + (float)k[p + 112] + (float)k[p + 113];
  float a2 = (float)v[p] + (float)v[p + 1] + (float)v[p + 112] + (float)v[p + 113];
  qg[idx] = (_Float16)(a0 * 0.25f);
  kg[idx] = (_Float16)(a1 * 0.25f);
  vg[idx] = (_Float16)(a2 * 0.25f);
}

// ------------------------------------------ patch scores: S[i,j] = sum_m q[m,i]k[m,j]
// grid (13,13,B), block 32; double-buffered async LDS staging of 32-row slabs
__global__ void k_pscore(const _Float16* __restrict__ q, const _Float16* __restrict__ k,
                         float* __restrict__ ps) {
  int b = blockIdx.z;
  int i0 = blockIdx.x * 16, j0 = blockIdx.y * 16;
  int lane = threadIdx.x, lm = lane & 15, l16 = lane >> 4;
  const _Float16* qb = q + (size_t)b * CHW;
  const _Float16* kb = k + (size_t)b * CHW;
  __shared__ __align__(16) _Float16 sq[2][32 * 16];
  __shared__ __align__(16) _Float16 sk[2][32 * 16];

  auto stage = [&](int buf, int k0) {
    const _Float16* qr = qb + (size_t)(k0 + lane) * PA + i0;
    const _Float16* kr = kb + (size_t)(k0 + lane) * PA + j0;
#if USE_ASYNC_LDS
#pragma unroll
    for (int c = 0; c < 4; c++) {
      __builtin_amdgcn_global_load_async_to_lds_b64(
          (g_v2i*)(gas_t*)(void*)(qr + c * 4),
          (l_v2i*)(las_t*)&sq[buf][lane * 16 + c * 4], 0, 0);
      __builtin_amdgcn_global_load_async_to_lds_b64(
          (g_v2i*)(gas_t*)(void*)(kr + c * 4),
          (l_v2i*)(las_t*)&sk[buf][lane * 16 + c * 4], 0, 0);
    }
#else
#pragma unroll
    for (int c = 0; c < 4; c++) {
      *(v4h*)&sq[buf][lane * 16 + c * 4] = *(const v4h*)(qr + c * 4);
      *(v4h*)&sk[buf][lane * 16 + c * 4] = *(const v4h*)(kr + c * 4);
    }
#endif
  };

  stage(0, 0);
  v8f acc = {};
  for (int k0 = 0; k0 < MROWS; k0 += 32) {
    int cur = (k0 >> 5) & 1;
    if (k0 + 32 < MROWS) {
      stage(cur ^ 1, k0 + 32);
#if USE_ASYNC_LDS
      __builtin_amdgcn_s_wait_asynccnt(8);  // oldest 8 (current slab) complete
#endif
    } else {
#if USE_ASYNC_LDS
      __builtin_amdgcn_s_wait_asynccnt(0);
#endif
    }
    v16h Af, Bf;
#pragma unroll
    for (int e = 0; e < 16; e++) {
      int kk = kof(e, l16);
      Af[e] = sq[cur][kk * 16 + lm];
      Bf[e] = sk[cur][kk * 16 + lm];
    }
    acc = wmma16(Af, Bf, acc);
  }
  const float scale = 0.015625f;  // 4096^-0.5
#pragma unroll
  for (int r = 0; r < 8; r++) {
    int ii = i0 + r + (l16 ? 8 : 0);
    ps[(size_t)b * PS_ROWS * PS_LD + (size_t)ii * PS_LD + j0 + lm] = acc[r] * scale;
  }
}

// ------------------------------------------------------------- patch softmax (rows)
__global__ void k_psoftmax(const float* __restrict__ ps, _Float16* __restrict__ pp) {
  int i = blockIdx.x, b = blockIdx.y, t = threadIdx.x;
  const float* row = ps + (size_t)b * PS_ROWS * PS_LD + (size_t)i * PS_LD;
  _Float16* orow   = pp + (size_t)b * PS_ROWS * PS_LD + (size_t)i * PS_LD;
  if (i >= PA) {
    for (int j = t; j < PS_LD; j += blockDim.x) orow[j] = (_Float16)0.f;
    return;
  }
  __shared__ float sh[256];
  float mx = -1e30f;
  for (int j = t; j < PA; j += blockDim.x) mx = fmaxf(mx, row[j]);
  sh[t] = mx; __syncthreads();
  for (int o = 128; o > 0; o >>= 1) { if (t < o) sh[t] = fmaxf(sh[t], sh[t + o]); __syncthreads(); }
  mx = sh[0]; __syncthreads();
  float s = 0.f;
  for (int j = t; j < PA; j += blockDim.x) s += __expf(row[j] - mx);
  sh[t] = s; __syncthreads();
  for (int o = 128; o > 0; o >>= 1) { if (t < o) sh[t] += sh[t + o]; __syncthreads(); }
  float inv = 1.f / sh[0];
  for (int j = t; j < PS_LD; j += blockDim.x)
    orow[j] = (j < PA) ? (_Float16)(__expf(row[j] - mx) * inv) : (_Float16)0.f;
}

// ----------------------------------- patch out: h[m,i] = sum_j v[m,j] * w[i,j]
// grid (256, 13, B), block 32
__global__ void k_phout(const _Float16* __restrict__ v, const _Float16* __restrict__ pp,
                        float* __restrict__ hpatch) {
  int b = blockIdx.z;
  int m0 = blockIdx.x * 16, i0 = blockIdx.y * 16;
  int lane = threadIdx.x, lm = lane & 15, l16 = lane >> 4;
  const _Float16* vb = v + (size_t)b * CHW;
  const _Float16* brow = pp + (size_t)b * PS_ROWS * PS_LD + (size_t)(i0 + lm) * PS_LD + l16 * 8;
  v8f acc = {};
  for (int k0 = 0; k0 < PS_LD; k0 += 32) {
    const v8h* bp = (const v8h*)(brow + k0);
    v16h Bf = CAT16(bp[0], bp[2]);
    v16h Af;
#pragma unroll
    for (int e = 0; e < 16; e++) {
      int j = k0 + kof(e, l16);
      Af[e] = (j < PA) ? vb[(size_t)(m0 + lm) * PA + j] : (_Float16)0.f;
    }
    acc = wmma16(Af, Bf, acc);
  }
#pragma unroll
  for (int r = 0; r < 8; r++) {
    int m = m0 + r + (l16 ? 8 : 0);
    int i = i0 + lm;
    if (i < PA) hpatch[(size_t)b * CHW + (size_t)m * PA + i] = acc[r];
  }
}

// ----------------------------- global scores (one batch): S[i,j]=sum_c qg[c,i]kg[c,j]
// grid (196,196), block 32
__global__ void k_gscore(const _Float16* __restrict__ qg, const _Float16* __restrict__ kg,
                         _Float16* __restrict__ gs) {
  int i0 = blockIdx.x * 16, j0 = blockIdx.y * 16;
  int lane = threadIdx.x, lm = lane & 15, l16 = lane >> 4;
  v8f acc = {};
#pragma unroll
  for (int k0 = 0; k0 < C64; k0 += 32) {
    v16h Af, Bf;
#pragma unroll
    for (int e = 0; e < 16; e++) {
      int c = k0 + kof(e, l16);
      Af[e] = qg[(size_t)c * NG + i0 + lm];
      Bf[e] = kg[(size_t)c * NG + j0 + lm];
    }
    acc = wmma16(Af, Bf, acc);
  }
  const float scale = 0.125f;  // 64^-0.5
#pragma unroll
  for (int r = 0; r < 8; r++) {
    int ii = i0 + r + (l16 ? 8 : 0);
    gs[(size_t)ii * NG + j0 + lm] = (_Float16)(acc[r] * scale);
  }
}

// --------------------------------------------- global softmax, in-place f16 rows
__global__ void k_gsoftmax(_Float16* __restrict__ gs) {
  int i = blockIdx.x, t = threadIdx.x;
  _Float16* row = gs + (size_t)i * NG;
  __shared__ float sh[256];
  float mx = -1e30f;
  for (int j = t; j < NG; j += blockDim.x) mx = fmaxf(mx, (float)row[j]);
  sh[t] = mx; __syncthreads();
  for (int o = 128; o > 0; o >>= 1) { if (t < o) sh[t] = fmaxf(sh[t], sh[t + o]); __syncthreads(); }
  mx = sh[0]; __syncthreads();
  float s = 0.f;
  for (int j = t; j < NG; j += blockDim.x) s += __expf((float)row[j] - mx);
  sh[t] = s; __syncthreads();
  for (int o = 128; o > 0; o >>= 1) { if (t < o) sh[t] += sh[t + o]; __syncthreads(); }
  float inv = 1.f / sh[0];
  for (int j = t; j < NG; j += blockDim.x)
    row[j] = (_Float16)(__expf((float)row[j] - mx) * inv);
}

// ---------------------- global out (one batch): h[c,i] = sum_j vg[c,j] * w[i,j]
// grid (196), block 128; fragments via aligned b128 pair loads
__global__ void k_gout(const _Float16* __restrict__ vg, const _Float16* __restrict__ gs,
                       float* __restrict__ hg) {
  int i0 = blockIdx.x * 16;
  int wave = threadIdx.x >> 5;
  int lane = threadIdx.x & 31, lm = lane & 15, l16 = lane >> 4;
  const _Float16* arow = vg + (size_t)(wave * 16 + lm) * NG + l16 * 8;
  const _Float16* brow = gs + (size_t)(i0 + lm) * NG + l16 * 8;
  v8f acc = {};
  for (int k0 = 0; k0 < NG; k0 += 32) {
    __builtin_prefetch(brow + k0 + 256, 0, 1);
    const v8h* ap = (const v8h*)(arow + k0);
    const v8h* bp = (const v8h*)(brow + k0);
    v16h Af = CAT16(ap[0], ap[2]);
    v16h Bf = CAT16(bp[0], bp[2]);
    acc = wmma16(Af, Bf, acc);
  }
#pragma unroll
  for (int r = 0; r < 8; r++) {
    int c = wave * 16 + r + (l16 ? 8 : 0);
    hg[(size_t)c * NG + i0 + lm] = acc[r];
  }
}

// ------------- final: out = x + Wproj @ (0.75*h_patch + 0.25*upsample(h_global))
// grid (784, B), block 128
__global__ void k_final(const float* __restrict__ x, const float* __restrict__ wproj,
                        const float* __restrict__ hpatch, const float* __restrict__ hg,
                        float* __restrict__ out) {
  int b = blockIdx.y;
  int s0 = blockIdx.x * 16;
  __shared__ __align__(16) _Float16 lw[C64 * C64];
  __shared__ __align__(32) _Float16 lh[2][32 * 16];  // swizzled fragment order
  for (int i = threadIdx.x; i < C64 * C64; i += blockDim.x)
    lw[i] = (_Float16)wproj[i];
  for (int i = threadIdx.x; i < C64 * 16; i += blockDim.x) {
    int c = i >> 4, n = i & 15;
    int s = s0 + n;
    float hp = hpatch[(size_t)b * CHW + (size_t)c * HW + s];
    int y = s / 112, xx = s % 112;
    float sy = y * 0.5f - 0.25f, sx = xx * 0.5f - 0.25f;
    int y0 = (int)floorf(sy), x0 = (int)floorf(sx);
    float fy = sy - (float)y0, fx = sx - (float)x0;
    int y0c = iclamp(y0, 0, TGW - 1), y1c = iclamp(y0 + 1, 0, TGW - 1);
    int x0c = iclamp(x0, 0, TGW - 1), x1c = iclamp(x0 + 1, 0, TGW - 1);
    const float* hc = hg + (size_t)b * C64 * NG + (size_t)c * NG;
    float v00 = hc[y0c * TGW + x0c], v01 = hc[y0c * TGW + x1c];
    float v10 = hc[y1c * TGW + x0c], v11 = hc[y1c * TGW + x1c];
    float hgv = (1.f - fy) * ((1.f - fx) * v00 + fx * v01) +
                fy * ((1.f - fx) * v10 + fx * v11);
    lh[c >> 5][swz_idx(c & 31, n)] = (_Float16)(0.75f * hp + 0.25f * hgv);
  }
  __syncthreads();

  int wave = threadIdx.x >> 5;
  int lane = threadIdx.x & 31, lm = lane & 15, l16 = lane >> 4;
  v8f acc = {};
#pragma unroll
  for (int ks = 0; ks < 2; ks++) {
    const v8h* ap = (const v8h*)&lw[(wave * 16 + lm) * C64 + ks * 32 + l16 * 8];
    v16h Af = CAT16(ap[0], ap[2]);
    v16h Bf = *(const v16h*)&lh[ks][lane * 16];
    acc = wmma16(Af, Bf, acc);
  }
#pragma unroll
  for (int r = 0; r < 8; r++) {
    int o = wave * 16 + r + (l16 ? 8 : 0);
    size_t idx = (size_t)b * CHW + (size_t)o * HW + s0 + lm;
    out[idx] = x[idx] + acc[r];
  }
}

// ---------------------------------------------------------------------- launcher
extern "C" void kernel_launch(void* const* d_in, const int* in_sizes, int n_in,
                              void* d_out, int out_size, void* d_ws, size_t ws_size,
                              hipStream_t stream) {
  const float* x     = (const float*)d_in[0];
  const float* gn_w  = (const float*)d_in[1];
  const float* gn_b  = (const float*)d_in[2];
  const float* wq    = (const float*)d_in[3];
  const float* bq    = (const float*)d_in[4];
  const float* wk    = (const float*)d_in[5];
  const float* bk    = (const float*)d_in[6];
  const float* wv    = (const float*)d_in[7];
  const float* bv    = (const float*)d_in[8];
  const float* wproj = (const float*)d_in[9];
  float* out = (float*)d_out;

  char* ws = (char*)d_ws;
  size_t off = 0;
  auto alloc = [&](size_t bytes) -> char* {
    char* p = ws + off;
    off += (bytes + 255) & ~(size_t)255;
    return p;
  };
  float*    stats  = (float*)   alloc(BB * 2 * sizeof(float));
  _Float16* qh     = (_Float16*)alloc((size_t)BB * CHW * 2 + 512);  // +tail pad
  _Float16* kh     = (_Float16*)alloc((size_t)BB * CHW * 2 + 512);
  _Float16* vh     = (_Float16*)alloc((size_t)BB * CHW * 2 + 512);
  _Float16* qg     = (_Float16*)alloc((size_t)BB * C64 * NG * 2);
  _Float16* kg     = (_Float16*)alloc((size_t)BB * C64 * NG * 2);
  _Float16* vg     = (_Float16*)alloc((size_t)BB * C64 * NG * 2);
  float*    ps     = (float*)   alloc((size_t)BB * PS_ROWS * PS_LD * 4);
  _Float16* pp     = (_Float16*)alloc((size_t)BB * PS_ROWS * PS_LD * 2);
  _Float16* gs     = (_Float16*)alloc((size_t)NG * NG * 2);   // one batch, reused
  float*    hg     = (float*)   alloc((size_t)BB * C64 * NG * 4);
  float*    hpatch = (float*)   alloc((size_t)BB * CHW * 4);
  (void)ws_size; (void)in_sizes; (void)n_in; (void)out_size;

  k_gnstats<<<BB, 256, 0, stream>>>(x, stats);
  k_qkv<<<dim3(HW / 64, BB), 128, 0, stream>>>(x, stats, gn_w, gn_b, wq, bq, wk,
                                               bk, wv, bv, qh, kh, vh);
  {
    int tot = BB * C64 * NG;
    k_pool<<<(tot + 255) / 256, 256, 0, stream>>>(qh, kh, vh, qg, kg, vg);
  }
  k_pscore<<<dim3(13, 13, BB), 32, 0, stream>>>(qh, kh, ps);
  k_psoftmax<<<dim3(PS_ROWS, BB), 256, 0, stream>>>(ps, pp);
  k_phout<<<dim3(MROWS / 16, 13, BB), 32, 0, stream>>>(vh, pp, hpatch);

  for (int b = 0; b < BB; b++) {
    const _Float16* qgb = qg + (size_t)b * C64 * NG;
    const _Float16* kgb = kg + (size_t)b * C64 * NG;
    const _Float16* vgb = vg + (size_t)b * C64 * NG;
    float* hgb = hg + (size_t)b * C64 * NG;
    k_gscore<<<dim3(NG / 16, NG / 16), 32, 0, stream>>>(qgb, kgb, gs);
    k_gsoftmax<<<NG, 256, 0, stream>>>(gs);
    k_gout<<<NG / 16, 128, 0, stream>>>(vgb, gs, hgb);
  }

  k_final<<<dim3(HW / 16, BB), 128, 0, stream>>>(x, wproj, hpatch, hg, out);
}
